// TritonMoEMLP_7937099563379
// MI455X (gfx1250) — matrix operand order repaired and checked by reading, not compiled, not run on hardware
//
#include <hip/hip_runtime.h>
#include <hip/hip_bf16.h>

// MoE config (static, mirrors reference)
#define T_TOK 2048
#define D_EMB 1024
#define N_EXP 8
#define TOT_W 12288

typedef __attribute__((ext_vector_type(16))) __bf16 v16bf;
typedef __attribute__((ext_vector_type(8)))  float  v8f;

union Frag32B { uint4 u[2]; v16bf v; };

// A fragment: 16 contiguous bf16 (elements 0..7 at p, 8..15 at p+8)
__device__ __forceinline__ void load_fragA(Frag32B& f, const unsigned short* p) {
  f.u[0] = *(const uint4*)(p);
  f.u[1] = *(const uint4*)(p + 8);
}
// B fragment: 16 contiguous bf16 (32 bytes)
__device__ __forceinline__ void load_fragB(Frag32B& f, const unsigned short* p) {
  const uint4* q = (const uint4*)p;
  f.u[0] = q[0];
  f.u[1] = q[1];
}

__device__ __forceinline__ unsigned short f32_to_bf16(float f) {
  unsigned int u = __float_as_uint(f);
  u += 0x7FFFu + ((u >> 16) & 1u);   // round-to-nearest-even
  return (unsigned short)(u >> 16);
}

// ---------------------------------------------------------------------------
// Kernel 0: zero output + expert counters
// ---------------------------------------------------------------------------
__global__ void zero_kernel(float* __restrict__ out, int* __restrict__ cnt) {
  int i = blockIdx.x * 256 + threadIdx.x;   // grid covers exactly T_TOK*D_EMB
  out[i] = 0.0f;
  if (i < N_EXP) cnt[i] = 0;
}

// ---------------------------------------------------------------------------
// Kernel 1: convert x (fp32) -> xb (bf16), 4 elems / thread
// ---------------------------------------------------------------------------
__global__ __launch_bounds__(256) void cvtx_kernel(const float* __restrict__ x,
                                                   unsigned short* __restrict__ xb) {
  int i = (blockIdx.x * 256 + threadIdx.x) * 4;
  float4 v = *(const float4*)(x + i);
  ushort4 o;
  o.x = f32_to_bf16(v.x); o.y = f32_to_bf16(v.y);
  o.z = f32_to_bf16(v.z); o.w = f32_to_bf16(v.w);
  *(ushort4*)(xb + i) = o;
}

// ---------------------------------------------------------------------------
// Kernel 2: tiled transpose + convert: dst[c*R + r] = bf16(src[r*C + c])
// 32x32 tile, 256 threads (32 x 8, 4 rows each). R,C multiples of 32.
// ---------------------------------------------------------------------------
__global__ __launch_bounds__(256) void transpose_cvt_kernel(
    const float* __restrict__ src, unsigned short* __restrict__ dst, int R, int C) {
  __shared__ unsigned short tile[32][33];
  int tc = blockIdx.x * 32, tr = blockIdx.y * 32;
  int tx = threadIdx.x & 31, tg = (threadIdx.x >> 5) * 4;
  #pragma unroll
  for (int j = 0; j < 4; j++)
    tile[tg + j][tx] = f32_to_bf16(src[(size_t)(tr + tg + j) * C + tc + tx]);
  __syncthreads();
  #pragma unroll
  for (int j = 0; j < 4; j++)
    dst[(size_t)(tc + tg + j) * R + tr + tx] = tile[tx][tg + j];
}

// ---------------------------------------------------------------------------
// Kernel 3: router — logits, sigmoid, top-2 (lowest-index tie-break like
// lax.top_k), renormalize, write combine[T][E], append token to expert lists.
// Output determinism: each out element later receives exactly TOP_K=2 fp32
// atomic adds; fp add of 2 values is commutative, so list order is irrelevant.
// ---------------------------------------------------------------------------
__global__ __launch_bounds__(256) void router_kernel(
    const float* __restrict__ x, const float* __restrict__ rw,
    float* __restrict__ combine, int* __restrict__ cnt, int* __restrict__ list) {
  int t = blockIdx.x * 256 + threadIdx.x;
  if (t >= T_TOK) return;
  const float* xr = x + (size_t)t * D_EMB;
  float acc[N_EXP];
  #pragma unroll
  for (int e = 0; e < N_EXP; e++) acc[e] = 0.0f;
  for (int i = 0; i < D_EMB; i++) {
    float xv = xr[i];
    #pragma unroll
    for (int e = 0; e < N_EXP; e++) acc[e] += xv * rw[e * D_EMB + i];
  }
  float p[N_EXP];
  #pragma unroll
  for (int e = 0; e < N_EXP; e++) p[e] = 1.0f / (1.0f + __expf(-acc[e]));
  int i0 = 0; float p0 = p[0];
  #pragma unroll
  for (int e = 1; e < N_EXP; e++) if (p[e] > p0) { p0 = p[e]; i0 = e; }
  int i1 = (i0 == 0) ? 1 : 0; float p1 = p[i1];
  #pragma unroll
  for (int e = 0; e < N_EXP; e++) if (e != i0 && p[e] > p1) { p1 = p[e]; i1 = e; }
  float s = 1.0f / (p0 + p1 + 1e-20f);
  float w0 = p0 * s, w1v = p1 * s;
  #pragma unroll
  for (int e = 0; e < N_EXP; e++)
    combine[t * N_EXP + e] = (e == i0) ? w0 : ((e == i1) ? w1v : 0.0f);
  int pos0 = atomicAdd(&cnt[i0], 1); list[i0 * T_TOK + pos0] = t;
  int pos1 = atomicAdd(&cnt[i1], 1); list[i1 * T_TOK + pos1] = t;
}

// ---------------------------------------------------------------------------
// Kernel 4: fused gathered expert MLP, bf16 WMMA, fp32 accumulate, TM=32.
// Block = 256 thr (8 waves), one expert e, 32 gathered tokens (2 M-halves).
// Each B fragment (single base reg + immediate offsets) feeds 2 WMMAs; L2
// latency is hidden across the 8 resident waves (compact ~210-VGPR footprint,
// no VGPR-MSB mode).
//   per 128-wide hidden chunk:
//     stage 1: H(32x128) = relu2( X(32x1024) @ W1t-chunk ), 2 tiles/wave
//     stage 2: acc(32x1024) += H @ W2t-chunk, 128 output cols/wave, 16 accs
//   epilogue: out[t, :] += combine[t,e] * acc (global_atomic_add_f32)
// ---------------------------------------------------------------------------
__global__ __launch_bounds__(256) void moe_wmma_kernel(
    const unsigned short* __restrict__ xb,    // [T][1024] bf16
    const unsigned short* __restrict__ w1t,   // [12288][1024] bf16 (w-major)
    const unsigned short* __restrict__ w2t,   // [1024][12288] bf16 (d-major)
    const float* __restrict__ combine,        // [T][8]
    const int*  __restrict__ cnt,             // [8]
    const int*  __restrict__ list,            // [8][T]
    float* __restrict__ out)                  // [T][1024]
{
  const int e    = blockIdx.y;
  const int tile = blockIdx.x;
  const int ce   = cnt[e];
  if (tile * 32 >= ce) return;                 // uniform early exit

  int off, wid;
  if (e < 4) { off = e << 10;                wid = 1024; }
  else       { off = 4096 + ((e - 4) << 11); wid = 2048; }
  const int nch = wid >> 7;                    // 128-wide chunks: 8 or 16

  __shared__ __align__(16) unsigned short Hs[32 * 128];  // chunk activations
  __shared__ int   toks[32];
  __shared__ float scl[32];

  const int tid  = threadIdx.x;
  const int wave = tid >> 5;
  const int lane = tid & 31;

  if (tid < 32) {
    int i = tile * 32 + tid;
    int t = list[e * T_TOK + (i < ce ? i : ce - 1)];
    toks[tid] = t;
    scl[tid]  = (i < ce) ? combine[t * N_EXP + e] : 0.0f;
  }
  __syncthreads();

  // WMMA fragment lane mapping (ISA 7.12.2, wave32)
  const int n  = lane & 15;            // B/D column within 16-tile; A row m
  const int kb = (lane >> 4) << 4;     // A: K half (0 or 16)
  const int kh = (lane >> 4) << 4;     // B: K half (0 or 16)
  const int mb = (lane >> 4) << 3;     // C/D: row base (0 or 8)

  const v8f vzero = {};
  v8f acc[16];
  #pragma unroll
  for (int j = 0; j < 16; j++) acc[j] = vzero;

  // A rows come straight from global bf16 x (WGP$/L2 resident, shared by waves)
  const unsigned short* xrow0 = xb + (size_t)toks[n]      * D_EMB + kb;
  const unsigned short* xrow1 = xb + (size_t)toks[16 + n] * D_EMB + kb;
  const unsigned short* hrow0 = &Hs[n * 128] + kb;
  const unsigned short* hrow1 = &Hs[(16 + n) * 128] + kb;

  for (int ch = 0; ch < nch; ch++) {
    // ---- stage 1: this wave computes H cols [wave*16, wave*16+16) ----
    v8f hc0 = vzero, hc1 = vzero;
    const unsigned short* w1col =
        w1t + (size_t)(off + ch * 128 + wave * 16 + n) * D_EMB + kh;
    if (ch + 1 < nch)                    // prefetch next chunk's W1 column
      __builtin_prefetch(w1col + (size_t)128 * D_EMB, 0, 1);
    #pragma unroll 4
    for (int kk = 0; kk < D_EMB; kk += 32) {
      Frag32B a0, a1, b;
      load_fragA(a0, xrow0 + kk);
      load_fragA(a1, xrow1 + kk);
      load_fragB(b, w1col + kk);
      hc0 = __builtin_amdgcn_wmma_f32_16x16x32_bf16(false, a0.v, false, b.v,
                                                    (short)0, hc0, false, false);
      hc1 = __builtin_amdgcn_wmma_f32_16x16x32_bf16(false, a1.v, false, b.v,
                                                    (short)0, hc1, false, false);
    }

    #pragma unroll
    for (int j = 0; j < 8; j++) {        // relu^2 + bf16, store to LDS
      float v0 = hc0[j];
      v0 = v0 > 0.0f ? v0 * v0 : 0.0f;
      Hs[(mb + j) * 128 + wave * 16 + n] = f32_to_bf16(v0);
      float v1 = hc1[j];
      v1 = v1 > 0.0f ? v1 * v1 : 0.0f;
      Hs[(16 + mb + j) * 128 + wave * 16 + n] = f32_to_bf16(v1);
    }
    __syncthreads();

    // ---- stage 2: this wave owns output cols [wave*128, wave*128+128) ----
    const int k2 = off + ch * 128;
    const unsigned short* w2base =
        w2t + (size_t)(wave * 128 + n) * TOT_W + k2 + kh;   // j stride: 16*TOT_W
    if (ch + 1 < nch) {                  // prefetch next chunk's W2 rows
      #pragma unroll
      for (int j = 0; j < 8; j++)
        __builtin_prefetch(w2base + (size_t)j * 16 * TOT_W + 128, 0, 1);
    }
    #pragma unroll
    for (int ks = 0; ks < 4; ks++) {
      const int kk = ks * 32;
      Frag32B a0, a1;
      load_fragA(a0, hrow0 + kk);
      load_fragA(a1, hrow1 + kk);
      #pragma unroll
      for (int j = 0; j < 8; j++) {
        Frag32B b;
        load_fragB(b, w2base + (size_t)j * 16 * TOT_W + kk);  // imm-offset stream
        acc[j] = __builtin_amdgcn_wmma_f32_16x16x32_bf16(false, a0.v, false, b.v,
                                                         (short)0, acc[j], false, false);
        acc[8 + j] = __builtin_amdgcn_wmma_f32_16x16x32_bf16(false, a1.v, false, b.v,
                                                             (short)0, acc[8 + j],
                                                             false, false);
      }
    }
    __syncthreads();   // Hs reused next chunk
  }

  // epilogue: scale by combine weight, atomic add (2 adds/element -> exact)
  #pragma unroll
  for (int j = 0; j < 8; j++) {
    const int ncol = wave * 128 + j * 16 + n;
    #pragma unroll
    for (int r = 0; r < 8; r++) {
      float s0 = scl[mb + r];
      if (s0 != 0.0f)
        unsafeAtomicAdd(out + (size_t)toks[mb + r] * D_EMB + ncol, acc[j][r] * s0);
      float s1 = scl[16 + mb + r];
      if (s1 != 0.0f)
        unsafeAtomicAdd(out + (size_t)toks[16 + mb + r] * D_EMB + ncol,
                        acc[8 + j][r] * s1);
    }
  }
}

// ---------------------------------------------------------------------------
// Host launcher. Workspace layout (needs ~54.7 MB):
//   combine 64KB | cnt 256B | list 64KB | xb 4MB | w1t 24MB | w2t 24MB
// ---------------------------------------------------------------------------
extern "C" void kernel_launch(void* const* d_in, const int* in_sizes, int n_in,
                              void* d_out, int out_size, void* d_ws, size_t ws_size,
                              hipStream_t stream) {
  (void)in_sizes; (void)n_in; (void)out_size; (void)ws_size;
  const float* x  = (const float*)d_in[0];
  const float* rw = (const float*)d_in[1];
  const float* w1 = (const float*)d_in[2];
  const float* w2 = (const float*)d_in[3];
  float* out = (float*)d_out;

  char* ws = (char*)d_ws;
  size_t cur = 0;
  float* combine = (float*)(ws + cur); cur += (size_t)T_TOK * N_EXP * 4;  // 65536
  int*   cnt     = (int*)(ws + cur);   cur += 256;
  int*   list    = (int*)(ws + cur);   cur += (size_t)N_EXP * T_TOK * 4;  // 65536
  unsigned short* xb  = (unsigned short*)(ws + cur); cur += (size_t)T_TOK * D_EMB * 2;
  unsigned short* w1t = (unsigned short*)(ws + cur); cur += (size_t)TOT_W * D_EMB * 2;
  unsigned short* w2t = (unsigned short*)(ws + cur); cur += (size_t)D_EMB * TOT_W * 2;

  zero_kernel<<<(T_TOK * D_EMB) / 256, 256, 0, stream>>>(out, cnt);
  cvtx_kernel<<<(T_TOK * D_EMB / 4) / 256, 256, 0, stream>>>(x, xb);
  // w1 [1024][12288] -> w1t [12288][1024]
  transpose_cvt_kernel<<<dim3(TOT_W / 32, D_EMB / 32), 256, 0, stream>>>(w1, w1t, D_EMB, TOT_W);
  // w2 [12288][1024] -> w2t [1024][12288]
  transpose_cvt_kernel<<<dim3(D_EMB / 32, TOT_W / 32), 256, 0, stream>>>(w2, w2t, TOT_W, D_EMB);
  router_kernel<<<T_TOK / 256, 256, 0, stream>>>(x, rw, combine, cnt, list);
  moe_wmma_kernel<<<dim3(T_TOK / 32, N_EXP), 256, 0, stream>>>(xb, w1t, w2t, combine,
                                                               cnt, list, out);
}
